// Decoder_4363686773371
// MI455X (gfx1250) — matrix operand order repaired and checked by reading, not compiled, not run on hardware
//
#include <hip/hip_runtime.h>
#include <hip/hip_bf16.h>

// ---------------------------------------------------------------------------
// LSTM decoder for MI455X (gfx1250, wave32).
// WMMA bf16 everywhere; double-buffered GLOBAL_LOAD_ASYNC_TO_LDS staging with
// partial s_wait_asynccnt so global latency hides behind the WMMA pipe.
// ---------------------------------------------------------------------------

typedef __attribute__((ext_vector_type(16))) __bf16 v16bf;
typedef __attribute__((ext_vector_type(8)))  float  v8f;
typedef __attribute__((ext_vector_type(8)))  __bf16 bf16x8;

#define Z_DIM 256
#define D_EMB 512
#define HID   1024
#define VOCAB 32000
#define BATCH 32
#define SEQT  128

__device__ __forceinline__ float sigmoidf_(float x) {
    return 1.0f / (1.0f + __expf(-x));
}

// Low 32 bits of a generic pointer into LDS == workgroup-relative LDS address.
__device__ __forceinline__ unsigned lds_addr32(const void* p) {
    return (unsigned)(size_t)p;
}

// Async copy global -> LDS (ASYNCcnt-tracked).
__device__ __forceinline__ void async_cp_b128(unsigned lds_off, const void* gaddr) {
    asm volatile("global_load_async_to_lds_b128 %0, %1, off"
                 :: "v"(lds_off), "v"(gaddr) : "memory");
}
__device__ __forceinline__ void async_cp_b64(unsigned lds_off, const void* gaddr) {
    asm volatile("global_load_async_to_lds_b64 %0, %1, off"
                 :: "v"(lds_off), "v"(gaddr) : "memory");
}
// Partial waits: async ops complete in order, so waiting ASYNCcnt<=N leaves
// the N newest copies (next slab) in flight while older ones are guaranteed.
__device__ __forceinline__ void wait_async0() {
    asm volatile("s_wait_asynccnt 0x0" ::: "memory");
}
__device__ __forceinline__ void wait_async1() {
    asm volatile("s_wait_asynccnt 0x1" ::: "memory");
}
__device__ __forceinline__ void wait_async2() {
    asm volatile("s_wait_asynccnt 0x2" ::: "memory");
}

__device__ __forceinline__ v16bf cat8(bf16x8 lo, bf16x8 hi) {
    return __builtin_shufflevector(lo, hi, 0, 1, 2, 3, 4, 5, 6, 7,
                                           8, 9, 10, 11, 12, 13, 14, 15);
}

// --- CDNA5 WMMA 16-bit A fragment (16x32, M x K) from an LDS tile ----------
// Per-lane data = one contiguous 16-elem run (k = 16h + s) -> 2x ds_load_b128.
__device__ __forceinline__ v16bf frag_a_lds(const __bf16* s, int row0, int ld) {
    int lane = threadIdx.x & 31;
    int m    = row0 + (lane & 15);
    int kofs = (lane & 16) ? 16 : 0;
    bf16x8 lo = *(const bf16x8*)&s[m * ld + kofs];
    bf16x8 hi = *(const bf16x8*)&s[m * ld + kofs + 8];
    return cat8(lo, hi);
}

// --- CDNA5 WMMA 16-bit B fragment (32x16, K x N) ---------------------------
// Tile stored N-MAJOR [n][k] (same layout as global W[N,K]): contiguous run.
__device__ __forceinline__ v16bf frag_b_nmajor(const __bf16* s, int col0, int ld) {
    int lane = threadIdx.x & 31;
    int n    = col0 + (lane & 15);
    int kofs = (lane & 16) ? 16 : 0;
    bf16x8 lo = *(const bf16x8*)&s[n * ld + kofs];
    bf16x8 hi = *(const bf16x8*)&s[n * ld + kofs + 8];
    return cat8(lo, hi);
}

// ---------------------------------------------------------------------------
// Generic bf16 GEMM: C[M,N] = A[M,K] * Bw[N,K]^T + bias[N]   (row-major)
// 128x128 tile per block, 256 threads = 8 wave32 waves, each wave 64x32.
// Double-buffered async-to-LDS pipeline.
// ---------------------------------------------------------------------------
__global__ __launch_bounds__(256) void k_gemm_bf16(
    const __bf16* __restrict__ A, const __bf16* __restrict__ Bw,
    const float* __restrict__ bias, float* __restrict__ C,
    int M, int N, int K)
{
    __shared__ __bf16 As[2][128 * 40];   // 128 M-rows x (32 K + 8 pad), x2
    __shared__ __bf16 Bs[2][128 * 40];   // 128 N-rows x (32 K + 8 pad), N-major

    const int tid = threadIdx.x;
    const int m0 = blockIdx.y * 128, n0 = blockIdx.x * 128;
    const int w  = tid >> 5;
    const int wm = (w & 1) * 64, wn = (w >> 1) * 32;

    v8f acc[4][2] = {};

    // Each thread stages 16B (8 bf16) at the start of one row-half of A and of B per slab.
    const int srow = tid >> 1, scb = (tid & 1) * 16;
    const unsigned ldsA[2] = { lds_addr32(&As[0][srow * 40 + scb]),
                               lds_addr32(&As[1][srow * 40 + scb]) };
    const unsigned ldsB[2] = { lds_addr32(&Bs[0][srow * 40 + scb]),
                               lds_addr32(&Bs[1][srow * 40 + scb]) };
    const __bf16* pA = A  + (size_t)(m0 + srow) * K + scb;
    const __bf16* pB = Bw + (size_t)(n0 + srow) * K + scb;

    // Prologue: stage slab 0.
    async_cp_b128(ldsA[0], pA);
    async_cp_b128(ldsB[0], pB);

    const int nk = K >> 5;
    for (int kb = 0; kb < nk; ++kb) {
        const int cur = kb & 1;
        if (kb + 1 < nk) {
            async_cp_b128(ldsA[cur ^ 1], pA + (kb + 1) * 32);
            async_cp_b128(ldsB[cur ^ 1], pB + (kb + 1) * 32);
            if (kb + 2 < nk) {  // global_prefetch_b8 slab kb+2
                __builtin_prefetch(pA + (kb + 2) * 32, 0, 1);
                __builtin_prefetch(pB + (kb + 2) * 32, 0, 1);
            }
            wait_async2();      // slab kb landed; kb+1 still in flight
        } else {
            wait_async0();
        }
        __syncthreads();

        v16bf bfr[2];
#pragma unroll
        for (int nt = 0; nt < 2; ++nt)
            bfr[nt] = frag_b_nmajor(Bs[cur], wn + 16 * nt, 40);
#pragma unroll
        for (int mt = 0; mt < 4; ++mt) {
            v16bf a = frag_a_lds(As[cur], wm + 16 * mt, 40);
#pragma unroll
            for (int nt = 0; nt < 2; ++nt)
                acc[mt][nt] = __builtin_amdgcn_wmma_f32_16x16x32_bf16(
                    false, a, false, bfr[nt], (short)0, acc[mt][nt], false, false);
        }
        __syncthreads();        // done reading buf 'cur' before it is refilled
    }

    // Epilogue per C/D layout: VGPR r -> M = r (+8 for hi lanes), N = lane&15.
    const int lane   = tid & 31;
    const int nlocal = lane & 15;
    const int mhi    = (lane & 16) ? 8 : 0;
#pragma unroll
    for (int mt = 0; mt < 4; ++mt) {
#pragma unroll
        for (int nt = 0; nt < 2; ++nt) {
            int col   = n0 + wn + 16 * nt + nlocal;
            int mbase = m0 + wm + 16 * mt + mhi;
            float bv  = bias ? bias[col] : 0.0f;
#pragma unroll
            for (int r = 0; r < 8; ++r)
                C[(size_t)(mbase + r) * N + col] = acc[mt][nt][r] + bv;
        }
    }
}

// ---------------------------------------------------------------------------
// Fused LSTM step: block j owns h-columns [64j, 64j+64).
// 32x64 slice of all 4 gates via WMMA (K=1024), + x_gates, pointwise update.
// ---------------------------------------------------------------------------
__global__ __launch_bounds__(256) void k_step(
    const float*  __restrict__ xg,    // [B*T, 4H]
    const __bf16* __restrict__ Whh,   // [4H, H] bf16
    const __bf16* __restrict__ h_in,  // [B, H]
    __bf16*       __restrict__ h_out, // [B, H]
    float*        __restrict__ c,     // [B, H]
    __bf16*       __restrict__ hseq,  // [B*T, H]
    int t)
{
    __shared__ __bf16 hS[2][32 * 40];     // 32 x 32 K-chunk of h, double buffer
    __shared__ float  Gs[4 * 32 * 64];    // gate slices i,f,g,o

    const int tid  = threadIdx.x;
    const int cb   = blockIdx.x * 64;
    const int w    = tid >> 5, lane = tid & 31;
    const int q    = w >> 1;              // gate 0..3
    const int mh   = w & 1;               // batch half

    v8f acc[4] = {};

    const int lr = tid >> 3, lk = (tid & 7) * 4;   // 4 bf16 = 8B per thread
    const unsigned ldsH[2] = { lds_addr32(&hS[0][lr * 40 + lk]),
                               lds_addr32(&hS[1][lr * 40 + lk]) };
    const __bf16* pH = h_in + lr * HID + lk;

    async_cp_b64(ldsH[0], pH);

    const int kofs = (lane & 16) ? 16 : 0;
    const int nk = HID >> 5;
    for (int kb = 0; kb < nk; ++kb) {
        const int cur = kb & 1;
        if (kb + 1 < nk) {
            async_cp_b64(ldsH[cur ^ 1], pH + (kb + 1) * 32);
            wait_async1();
        } else {
            wait_async0();
        }
        __syncthreads();

        v16bf a = frag_a_lds(hS[cur], 16 * mh, 40);
        const int k0 = kb * 32;
#pragma unroll
        for (int nt = 0; nt < 4; ++nt) {
            int nrow = q * HID + cb + nt * 16 + (lane & 15);
            const __bf16* src = &Whh[(size_t)nrow * HID + k0 + kofs];
            v16bf b = cat8(*(const bf16x8*)src, *(const bf16x8*)(src + 8));
            acc[nt] = __builtin_amdgcn_wmma_f32_16x16x32_bf16(
                false, a, false, b, (short)0, acc[nt], false, false);
        }
        __syncthreads();
    }

    const int nlocal = lane & 15;
    const int mhi    = (lane & 16) ? 8 : 0;
#pragma unroll
    for (int nt = 0; nt < 4; ++nt) {
#pragma unroll
        for (int r = 0; r < 8; ++r) {
            int m   = 16 * mh + mhi + r;      // batch index
            int col = nt * 16 + nlocal;       // 0..63
            float v = acc[nt][r] +
                      xg[(size_t)(m * SEQT + t) * (4 * HID) + q * HID + cb + col];
            Gs[(q * 32 + m) * 64 + col] = v;
        }
    }
    __syncthreads();

#pragma unroll
    for (int i = 0; i < 8; ++i) {
        int e   = i * 256 + tid;
        int m   = e >> 6, col = e & 63;
        int hc  = cb + col;
        float ig = sigmoidf_(Gs[(0 * 32 + m) * 64 + col]);
        float fg = sigmoidf_(Gs[(1 * 32 + m) * 64 + col]);
        float gg = tanhf    (Gs[(2 * 32 + m) * 64 + col]);
        float og = sigmoidf_(Gs[(3 * 32 + m) * 64 + col]);
        float cn = fg * c[m * HID + hc] + ig * gg;
        c[m * HID + hc] = cn;
        float hv = og * tanhf(cn);
        __bf16 hb = (__bf16)hv;
        h_out[m * HID + hc] = hb;
        hseq[(size_t)(m * SEQT + t) * HID + hc] = hb;
    }
}

// ---------------------------------------------------------------------------
// Small helper kernels
// ---------------------------------------------------------------------------
__global__ void k_h0(const float* __restrict__ z, const float* __restrict__ Wh,
                     const float* __restrict__ bh, __bf16* __restrict__ h0,
                     float* __restrict__ c)
{
    int g = blockIdx.x * blockDim.x + threadIdx.x;  // 32768
    int b = g >> 10, j = g & 1023;
    const float* zr = z + b * Z_DIM;
    const float* wr = Wh + j * Z_DIM;
    float s = 0.0f;
    for (int k = 0; k < Z_DIM; ++k) s += zr[k] * wr[k];
    h0[g] = (__bf16)tanhf(s + bh[j]);
    c[g]  = 0.0f;
}

__global__ void k_embed(const int* __restrict__ x, const float* __restrict__ emb,
                        __bf16* __restrict__ e)
{
    int row = blockIdx.x;                 // b*T + t
    int tok = x[row];
    const float* src = emb + (size_t)tok * D_EMB;
    __bf16*      dst = e   + (size_t)row * D_EMB;
    for (int i = threadIdx.x; i < D_EMB; i += blockDim.x)
        dst[i] = (__bf16)src[i];
}

__global__ void k_f2bf(const float* __restrict__ in, __bf16* __restrict__ out,
                       size_t n)
{
    size_t i = (size_t)blockIdx.x * blockDim.x + threadIdx.x;
    if (i < n) out[i] = (__bf16)in[i];
}

__global__ void k_addbias(const float* __restrict__ a, const float* __restrict__ b,
                          float* __restrict__ o, int n)
{
    int i = blockIdx.x * blockDim.x + threadIdx.x;
    if (i < n) o[i] = a[i] + b[i];
}

// ---------------------------------------------------------------------------
extern "C" void kernel_launch(void* const* d_in, const int* in_sizes, int n_in,
                              void* d_out, int out_size, void* d_ws, size_t ws_size,
                              hipStream_t stream)
{
    const float* z     = (const float*)d_in[0];
    const int*   x     = (const int*)  d_in[1];
    const float* W_h   = (const float*)d_in[2];
    const float* b_h   = (const float*)d_in[3];
    const float* emb   = (const float*)d_in[4];
    const float* W_ih  = (const float*)d_in[5];
    const float* W_hh  = (const float*)d_in[6];
    const float* b_ih  = (const float*)d_in[7];
    const float* b_hh  = (const float*)d_in[8];
    const float* W_out = (const float*)d_in[9];
    const float* b_out = (const float*)d_in[10];
    float* out = (float*)d_out;

    char* p = (char*)d_ws;
    auto alloc = [&](size_t bytes) -> void* {
        void* r = (void*)p;
        p += (bytes + 255) & ~(size_t)255;
        return r;
    };
    __bf16* Wih_bf  = (__bf16*)alloc((size_t)4 * HID * D_EMB * 2);
    __bf16* Whh_bf  = (__bf16*)alloc((size_t)4 * HID * HID * 2);
    __bf16* Wout_bf = (__bf16*)alloc((size_t)VOCAB * HID * 2);
    __bf16* e_bf    = (__bf16*)alloc((size_t)BATCH * SEQT * D_EMB * 2);
    float*  xg      = (float*) alloc((size_t)BATCH * SEQT * 4 * HID * 4);
    float*  bias4   = (float*) alloc((size_t)4 * HID * 4);
    __bf16* hbuf    = (__bf16*)alloc((size_t)2 * BATCH * HID * 2);  // double buffer
    float*  c       = (float*) alloc((size_t)BATCH * HID * 4);
    __bf16* hseq    = (__bf16*)alloc((size_t)BATCH * SEQT * HID * 2);

    // 1) One-shot weight conversion fp32 -> bf16
    size_t n_ih  = (size_t)4 * HID * D_EMB;
    size_t n_hh  = (size_t)4 * HID * HID;
    size_t n_out = (size_t)VOCAB * HID;
    k_f2bf<<<(unsigned)((n_ih  + 255) / 256), 256, 0, stream>>>(W_ih,  Wih_bf,  n_ih);
    k_f2bf<<<(unsigned)((n_hh  + 255) / 256), 256, 0, stream>>>(W_hh,  Whh_bf,  n_hh);
    k_f2bf<<<(unsigned)((n_out + 255) / 256), 256, 0, stream>>>(W_out, Wout_bf, n_out);
    k_addbias<<<16, 256, 0, stream>>>(b_ih, b_hh, bias4, 4 * HID);

    // 2) h0 = tanh(z W_h^T + b_h); c = 0
    k_h0<<<(BATCH * HID) / 256, 256, 0, stream>>>(z, W_h, b_h, hbuf, c);

    // 3) embedding gather -> bf16
    k_embed<<<BATCH * SEQT, 256, 0, stream>>>(x, emb, e_bf);

    // 4) x_gates = e @ W_ih^T + (b_ih + b_hh)   [4096 x 4096], K=512
    {
        dim3 g((4 * HID) / 128, (BATCH * SEQT) / 128);  // 32 x 32
        k_gemm_bf16<<<g, 256, 0, stream>>>(e_bf, Wih_bf, bias4, xg,
                                           BATCH * SEQT, 4 * HID, D_EMB);
    }

    // 5) recurrence: 128 fused steps
    for (int t = 0; t < SEQT; ++t) {
        __bf16* hin  = hbuf + (size_t)(t & 1) * BATCH * HID;
        __bf16* hout = hbuf + (size_t)((t + 1) & 1) * BATCH * HID;
        k_step<<<HID / 64, 256, 0, stream>>>(xg, Whh_bf, hin, hout, c, hseq, t);
    }

    // 6) logits = hseq @ W_out^T + b_out   [4096 x 32000], K=1024
    {
        dim3 g(VOCAB / 128, (BATCH * SEQT) / 128);      // 250 x 32
        k_gemm_bf16<<<g, 256, 0, stream>>>(hseq, Wout_bf, b_out, out,
                                           BATCH * SEQT, VOCAB, HID);
    }
}
